// NB101Flow_33638183862929
// MI455X (gfx1250) — matrix operand order repaired and checked by reading, not compiled, not run on hardware
//
#include <hip/hip_runtime.h>

// ---------------------------------------------------------------------------
// NB101Flow scorer for gfx1250 (MI455X).
//  - setup kernels precompute: y0 row pattern, sigmoid gate table, f16 weights
//  - main kernel: 2 graphs per 16-row WMMA tile per wave
//      support = y @ W        -> v_wmma_f32_16x16x32_f16  (f16 A/B, f32 acc)
//      agg     = (adj+I) @ s  -> v_wmma_f32_16x16x4_f32   (full f32)
//      gate*agg + bias, relu  -> VALU (branchless epilogue), f16 LDS round-trip
//  - MLP tail (128->200->1) with f16 weights + ds_swizzle lane reduction
// ---------------------------------------------------------------------------

typedef _Float16 v16h __attribute__((ext_vector_type(16)));
typedef _Float16 v8h  __attribute__((ext_vector_type(8)));
typedef float    v8f  __attribute__((ext_vector_type(8)));
typedef float    v2f  __attribute__((ext_vector_type(2)));

#define SWZ(x, imm) __builtin_bit_cast(float, \
    __builtin_amdgcn_ds_swizzle(__builtin_bit_cast(int, (x)), (imm)))

static __device__ __forceinline__ v16h cat8(v8h lo, v8h hi) {
  return __builtin_shufflevector(lo, hi, 0,1,2,3,4,5,6,7,8,9,10,11,12,13,14,15);
}

// ---------------- d_ws layout (bytes) ----------------
// [0,      4096)   y0 pattern, f16 [16][128]
// [4096,  19456)   gate table, f32 [5][6][128]
// [20480, 184320)  gcn weights, f16 [5][128][128]  (layer0 K-padded 96->128)
// [184320,235520)  mlp_w1^T,  f16 [200][128]

__global__ __launch_bounds__(128)
void nb101_setup_small(const float* __restrict__ in_node, const float* __restrict__ oth_node,
                       const float* __restrict__ in_op,  const float* __restrict__ op_tab,
                       const float* __restrict__ out_op,
                       const float* __restrict__ xh_w,   const float* __restrict__ xh_b,
                       const float* __restrict__ attn_w, const float* __restrict__ attn_b,
                       _Float16* __restrict__ wsY0, float* __restrict__ wsGate) {
  const int h = threadIdx.x;  // 0..127
  // y0 rows: row0/8 = input-node, rows 1-6/9-14 = other-node, rows 7/15 = 0
  float vin = 0.f, voth = 0.f;
  if (h < 96) {
    vin = xh_b[h]; voth = xh_b[h];
    for (int d = 0; d < 48; ++d) {
      const float w = xh_w[d * 96 + h];
      vin  += in_node[d]  * w;
      voth += oth_node[d] * w;
    }
  }
  for (int r = 0; r < 16; ++r) {
    float v = (r == 7 || r == 15) ? 0.f : (((r & 7) == 0) ? vin : voth);
    wsY0[r * 128 + h] = (_Float16)v;
  }
  // gate table: sigmoid(op_emb_t @ attn_w[l] + attn_b[l]) for 6 op types
  for (int l = 0; l < 5; ++l) {
    for (int t = 0; t < 6; ++t) {
      const float* e = (t == 0) ? in_op : ((t <= 4) ? (op_tab + (t - 1) * 48) : out_op);
      float s = attn_b[l * 128 + h];
      for (int d = 0; d < 48; ++d) s += e[d] * attn_w[(l * 48 + d) * 128 + h];
      wsGate[(l * 6 + t) * 128 + h] = 1.f / (1.f + __expf(-s));
    }
  }
}

__global__ __launch_bounds__(256)
void nb101_setup_weights(const float* __restrict__ gcn0_w, const float* __restrict__ gcn_w,
                         const float* __restrict__ mlp_w1,
                         _Float16* __restrict__ wsW, _Float16* __restrict__ wsW1T) {
  const int i = blockIdx.x * 256 + threadIdx.x;
  if (i < 5 * 128 * 128) {
    const int l = i >> 14, k = (i >> 7) & 127, n = i & 127;
    float v = (l == 0) ? ((k < 96) ? gcn0_w[k * 128 + n] : 0.f)
                       : gcn_w[(l - 1) * 16384 + k * 128 + n];
    wsW[i] = (_Float16)v;
  }
  if (i < 200 * 128) {
    const int o = i >> 7, k = i & 127;
    wsW1T[i] = (_Float16)mlp_w1[k * 200 + o];
  }
}

__global__ __launch_bounds__(256)
void nb101_main(const float* __restrict__ adjs, const int* __restrict__ opind,
                const float* __restrict__ gcn_b,
                const float* __restrict__ mlp_b1, const float* __restrict__ mlp_w2,
                const float* __restrict__ mlp_b2,
                const _Float16* __restrict__ wsY0, const float* __restrict__ wsGate,
                const _Float16* __restrict__ wsW,  const _Float16* __restrict__ wsW1T,
                float* __restrict__ out, int npairs, int iters) {
  __shared__ __align__(16) _Float16 sW[128 * 128];     // current-layer weights, [k][n]
  __shared__ __align__(16) _Float16 sY0[16 * 128];     // initial activation pattern
  __shared__ __align__(16) _Float16 sAct[8][16 * 128]; // per-wave activation staging
  __shared__ float sGate[5 * 6 * 128];
  __shared__ float sBias[5 * 128];
  __shared__ float sYbar[8][2][128];

  const int tid  = threadIdx.x;
  const int wv   = tid >> 5;
  const int lane = tid & 31;
  const int m    = lane & 15;   // WMMA row / column-within-tile index
  const int hh   = lane >> 4;   // lane half: selects graph for C/D rows

  // one-time staging
  for (int i = tid; i < 256; i += 256) ((uint4*)sY0)[i] = ((const uint4*)wsY0)[i];
  for (int i = tid; i < 5 * 6 * 128; i += 256) sGate[i] = wsGate[i];
  for (int i = tid; i < 5 * 128; i += 256) sBias[i] = gcn_b[i];
  __syncthreads();

  const int stride = gridDim.x * 8;

  for (int it = 0; it < iters; ++it) {
    const int pair = blockIdx.x * 8 + wv + it * stride;
    const bool active = pair < npairs;

    if (pair + stride < npairs)
      __builtin_prefetch(adjs + (size_t)(pair + stride) * 2 * 49, 0, 1);

    // ---- gate type (per C/D-row j of my half's graph) + mean-mask weights ----
    int   tt[8];
    float mw[8];
    const int g = pair * 2 + hh;
    tt[0] = 0; mw[0] = 0.f;
    #pragma unroll
    for (int jj = 0; jj < 5; ++jj) {
      const int op = active ? opind[g * 5 + jj] : 0;
      tt[1 + jj] = 1 + op;
      mw[1 + jj] = (op != 3) ? 1.f : 0.f;
    }
    tt[6] = 5; mw[6] = 1.f;
    tt[7] = 0; mw[7] = 0.f;

    // ---- adjacency A-fragments for v_wmma_f32_16x16x4_f32 ----
    // A(16x16) = blockdiag(adj0+I, adj1+I); chunk kc: VGPR q holds K=4kc+2*hh+q
    float adjA[4][2];
    #pragma unroll
    for (int kc = 0; kc < 4; ++kc) {
      #pragma unroll
      for (int q = 0; q < 2; ++q) {
        const int k = 4 * kc + 2 * hh + q;
        const int r = m, c = k;
        float v = 0.f;
        const bool blk0 = (r < 7 && c < 7);
        const bool blk1 = (r >= 8 && r < 15 && c >= 8 && c < 15);
        if (active && (blk0 || blk1)) {
          const int gg = blk0 ? (pair * 2) : (pair * 2 + 1);
          const int rr = blk0 ? r : r - 8;
          const int cc = blk0 ? c : c - 8;
          v = adjs[(size_t)gg * 49 + rr * 7 + cc] + ((rr == cc) ? 1.f : 0.f);
        }
        adjA[kc][q] = v;
      }
    }

    // ---------------- 5 GCN layers ----------------
    for (int l = 0; l < 5; ++l) {
      __syncthreads();
      {
        const uint4* src = (const uint4*)(wsW + l * 16384);
        for (int i = tid; i < 2048; i += 256) ((uint4*)sW)[i] = src[i];
      }
      __syncthreads();

      if (active) {
        const _Float16* As = (l == 0) ? sY0 : sAct[wv];
        // A-fragments: lane m row; K set depends on lane half (16-bit A layout)
        v16h aA[4];
        #pragma unroll
        for (int c = 0; c < 4; ++c) {
          v8h lo = *(const v8h*)(As + m * 128 + 32 * c + 8 * hh);
          v8h hi = *(const v8h*)(As + m * 128 + 32 * c + 16 + 8 * hh);
          aA[c] = cat8(lo, hi);
        }

        #pragma unroll
        for (int t = 0; t < 8; ++t) {
          // support tile: acc[j] = row (j + 8*hh), col (m + 16*t)
          v8f acc = {0.f, 0.f, 0.f, 0.f, 0.f, 0.f, 0.f, 0.f};
          #pragma unroll
          for (int c = 0; c < 4; ++c) {
            v16h B;
            #pragma unroll
            for (int s = 0; s < 16; ++s)
              B[s] = sW[(32 * c + ((s < 8) ? (8 * hh + s) : (16 + 8 * hh + s - 8))) * 128 + 16 * t + m];
            acc = __builtin_amdgcn_wmma_f32_16x16x32_f16(
                false, aA[c], false, B, (short)0, acc, false, false);
          }

          // aggregation: agg = blockdiag(adj+I) @ support, in f32 WMMA
          v8f agg = {0.f, 0.f, 0.f, 0.f, 0.f, 0.f, 0.f, 0.f};
          #pragma unroll
          for (int kc = 0; kc < 4; ++kc) {
            v2f A2, B2;
            A2[0] = adjA[kc][0];
            A2[1] = adjA[kc][1];
            #pragma unroll
            for (int q = 0; q < 2; ++q) {
              const int j0 = (4 * kc + q) & 7;       // K for lanes 0-15
              const int j1 = (4 * kc + 2 + q) & 7;   // K for lanes 16-31
              const float d0 = acc[j0];
              const float d1 = acc[j1];
              const float sw = SWZ((kc < 2) ? d1 : d0, 0x401f);  // SWAPX16
              B2[q] = (kc < 2) ? ((hh == 0) ? d0 : sw)
                               : ((hh == 0) ? sw : d1);
            }
            agg = __builtin_amdgcn_wmma_f32_16x16x4_f32(
                false, A2, false, B2, (short)0, agg, false, false);
          }

          // Branchless epilogue:
          //  - relu'd f16 activation always stored (harmless extra on layer 4)
          //  - masked mean always accumulated with pre-relu value (only layer-4
          //    write of sYbar survives, it is the last one before the MLP reads)
          const int col = m + 16 * t;
          const float bias = sBias[l * 128 + col];
          const float* gl = sGate + l * 6 * 128;
          float ybp = 0.f;
          #pragma unroll
          for (int j = 0; j < 8; ++j) {
            const float v = gl[tt[j] * 128 + col] * agg[j] + bias;
            sAct[wv][(j + 8 * hh) * 128 + col] = (_Float16)fmaxf(v, 0.f);
            ybp += mw[j] * v;
          }
          sYbar[wv][hh][col] = ybp * (1.f / 6.f);
        }
      }
    }

    // ---------------- MLP tail: 128 -> 200 -> 1 ----------------
    if (active) {
      const float bb2 = mlp_b2[0];
      #pragma unroll
      for (int gg = 0; gg < 2; ++gg) {
        float hs = 0.f;
        const float4* yb4 = (const float4*)(&sYbar[wv][gg][0]);
        for (int o = lane; o < 200; o += 32) {
          float a = mlp_b1[o];
          const v8h* wrow = (const v8h*)(wsW1T + o * 128);
          #pragma unroll
          for (int kb = 0; kb < 16; ++kb) {
            const v8h w8 = wrow[kb];
            const float4 ya = yb4[2 * kb];
            const float4 yz = yb4[2 * kb + 1];
            a += (float)w8[0] * ya.x + (float)w8[1] * ya.y +
                 (float)w8[2] * ya.z + (float)w8[3] * ya.w +
                 (float)w8[4] * yz.x + (float)w8[5] * yz.y +
                 (float)w8[6] * yz.z + (float)w8[7] * yz.w;
          }
          hs += fmaxf(a, 0.f) * mlp_w2[o];
        }
        // full-wave reduction via ds_swizzle xor masks 1,2,4,8,16
        hs += SWZ(hs, 0x041f);
        hs += SWZ(hs, 0x081f);
        hs += SWZ(hs, 0x101f);
        hs += SWZ(hs, 0x201f);
        hs += SWZ(hs, 0x401f);
        if (lane == 0) out[pair * 2 + gg] = hs + bb2;
      }
    }
  }
}

extern "C" void kernel_launch(void* const* d_in, const int* in_sizes, int n_in,
                              void* d_out, int out_size, void* d_ws, size_t ws_size,
                              hipStream_t stream) {
  const float* adjs     = (const float*)d_in[0];
  const int*   op_inds  = (const int*)d_in[1];
  const float* in_node  = (const float*)d_in[2];
  const float* oth_node = (const float*)d_in[3];
  const float* in_op    = (const float*)d_in[4];
  const float* op_tab   = (const float*)d_in[5];
  const float* out_op   = (const float*)d_in[6];
  const float* xh_w     = (const float*)d_in[7];
  const float* xh_b     = (const float*)d_in[8];
  const float* gcn0_w   = (const float*)d_in[9];
  const float* gcn_w    = (const float*)d_in[10];
  const float* attn_w   = (const float*)d_in[11];
  const float* attn_b   = (const float*)d_in[12];
  const float* gcn_b    = (const float*)d_in[13];
  const float* mlp_w1   = (const float*)d_in[14];
  const float* mlp_b1   = (const float*)d_in[15];
  const float* mlp_w2   = (const float*)d_in[16];
  const float* mlp_b2   = (const float*)d_in[17];

  char* ws = (char*)d_ws;
  _Float16* wsY0  = (_Float16*)(ws + 0);
  float*    wsGate= (float*)(ws + 4096);
  _Float16* wsW   = (_Float16*)(ws + 20480);
  _Float16* wsW1T = (_Float16*)(ws + 20480 + 163840);

  const int B = in_sizes[0] / 49;
  const int npairs = B / 2;

  nb101_setup_small<<<1, 128, 0, stream>>>(in_node, oth_node, in_op, op_tab, out_op,
                                           xh_w, xh_b, attn_w, attn_b, wsY0, wsGate);
  nb101_setup_weights<<<320, 256, 0, stream>>>(gcn0_w, gcn_w, mlp_w1, wsW, wsW1T);

  const int nblocks = 512;
  const int totalWaves = nblocks * 8;
  const int iters = (npairs + totalWaves - 1) / totalWaves;
  nb101_main<<<nblocks, 256, 0, stream>>>(adjs, op_inds, gcn_b, mlp_b1, mlp_w2, mlp_b2,
                                          wsY0, wsGate, wsW, wsW1T,
                                          (float*)d_out, npairs, iters);
}
